// SpectralConv2d_fast_20143396618631
// MI455X (gfx1250) — matrix operand order repaired and checked by reading, not compiled, not run on hardware
//
#include <hip/hip_runtime.h>
#include <hip/hip_bf16.h>
#include <math.h>

// ---------------------------------------------------------------------------
// SpectralConv2d (FNO) for gfx1250: everything expressed as bf16 WMMA GEMMs.
// DFTs are done as matrix multiplies (128-pt DFT == 128x128 GEMM), complex
// arithmetic via interleaved activations + 2x2-block-expanded constant B.
// ASYNCcnt-tracked global->LDS staging + global prefetch.
// ---------------------------------------------------------------------------

typedef __attribute__((ext_vector_type(16))) __bf16 v16bf;
typedef __attribute__((ext_vector_type(8)))  float  v8f;
typedef int v4i_vs __attribute__((vector_size(16)));   // matches builtin param

#if defined(__gfx1250__) && __has_builtin(__builtin_amdgcn_global_load_async_to_lds_b128)
#define HAVE_ASYNC_COPY 1
#endif

__device__ inline void async_copy16(const void* g, void* l) {
#ifdef HAVE_ASYNC_COPY
  __builtin_amdgcn_global_load_async_to_lds_b128(
      (__attribute__((address_space(1))) v4i_vs*)(g),
      (__attribute__((address_space(3))) v4i_vs*)(l), 0, 0);
#endif
}
__device__ inline void async_wait0() {
#ifdef HAVE_ASYNC_COPY
#if __has_builtin(__builtin_amdgcn_s_wait_asynccnt)
  __builtin_amdgcn_s_wait_asynccnt(0);
#else
  asm volatile("s_wait_asynccnt 0" ::: "memory");
#endif
#endif
}

__device__ inline __bf16 f2bf(float f) {
  union { float f; unsigned u; } v; v.f = f;
  unsigned r = v.u + 0x7FFFu + ((v.u >> 16) & 1u);
  unsigned short h = (unsigned short)(r >> 16);
  __bf16 b; __builtin_memcpy(&b, &h, 2); return b;
}
__device__ inline float gelu_f(float x) {
  return 0.5f * x * (1.0f + erff(x * 0.70710678118654752440f));
}

// ---------------------------------------------------------------------------
// Constant generation: DFT matrices (bf16)
//  cwf [128][160]: forward rfft along W (real input), cols interleaved re/im, n>=65 zero
//  chf [256][256]: forward fft along H, expanded complex  [[r,i],[-i,r]]
//  chi [256][256]: inverse fft along H (x 1/128), expanded complex
//  b5  [160][128]: inverse rfft along W (x c_n/128), real output
// ---------------------------------------------------------------------------
__global__ __launch_bounds__(256) void gen_consts_k(__bf16* cwf, __bf16* chf,
                                                    __bf16* chi, __bf16* b5) {
  int i = blockIdx.x * 256 + threadIdx.x;
  const float TWO_PI = 6.283185307179586f;
  if (i < 20480) {                      // cwf
    int w = i / 160, j = i % 160, n = j >> 1, c = j & 1;
    float v = 0.f;
    if (n < 65) {
      float th = TWO_PI * (float)((w * n) & 127) * (1.f / 128.f);
      v = c ? -sinf(th) : cosf(th);
    }
    cwf[i] = f2bf(v);
  } else if (i < 86016) {               // chf
    int l = i - 20480; int row = l >> 8, col = l & 255;
    int h = row >> 1, a = row & 1, hp = col >> 1, b = col & 1;
    float th = TWO_PI * (float)((h * hp) & 127) * (1.f / 128.f);
    float Br = cosf(th), Bi = -sinf(th);
    float v = (a == b) ? Br : (a ? -Bi : Bi);
    chf[l] = f2bf(v);
  } else if (i < 151552) {              // chi
    int l = i - 86016; int row = l >> 8, col = l & 255;
    int h2 = row >> 1, a = row & 1, hh = col >> 1, b = col & 1;
    float th = TWO_PI * (float)((h2 * hh) & 127) * (1.f / 128.f);
    float Br = cosf(th) * (1.f / 128.f), Bi = sinf(th) * (1.f / 128.f);
    float v = (a == b) ? Br : (a ? -Bi : Bi);
    chi[l] = f2bf(v);
  } else if (i < 172032) {              // b5
    int l = i - 151552; int r = l >> 7, w = l & 127;
    int n = r >> 1, c = r & 1;
    float v = 0.f;
    if (n < 65) {
      float cn = (n == 0 || n == 64) ? 1.f : 2.f;
      float th = TWO_PI * (float)((n * w) & 127) * (1.f / 128.f);
      v = (c ? -cn * sinf(th) : cn * cosf(th)) * (1.f / 128.f);
    }
    b5[l] = f2bf(v);
  }
}

// ---------------------------------------------------------------------------
// MLP hidden stack: 2 -> 64 (gelu) -> 64 (gelu) -> 64 (gelu); per-point, LDS
// scratch rows. 8320 grid points, block = 64 threads (one point per thread).
// ---------------------------------------------------------------------------
__global__ __launch_bounds__(64) void mlp_hidden_k(
    const float* __restrict__ w_in, const float* __restrict__ b_in,
    const float* __restrict__ wh0, const float* __restrict__ wh1,
    const float* __restrict__ bh0, const float* __restrict__ bh1,
    __bf16* __restrict__ hid) {
  __shared__ float sa[64][65];
  __shared__ float sb[64][65];
  int t = threadIdx.x;
  int g = blockIdx.x * 64 + t;                 // 0..8319 (grid is exactly 130)
  int h = g / 65, n = g % 65;
  float x0 = ((h < 64) ? (float)h : (float)(h - 128)) * (1.f / 128.f);
  float x1 = (float)n * (1.f / 128.f);
  for (int j = 0; j < 64; ++j)
    sa[t][j] = gelu_f(x0 * w_in[j] + x1 * w_in[64 + j] + b_in[j]);
  const float* whs[2] = { wh0, wh1 };
  const float* bhs[2] = { bh0, bh1 };
  for (int L = 0; L < 2; ++L) {
    for (int j = 0; j < 64; ++j) {
      float s = bhs[L][j];
      for (int k = 0; k < 64; ++k) s += sa[t][k] * whs[L][k * 64 + j];
      sb[t][j] = gelu_f(s);
    }
    for (int j = 0; j < 64; ++j) sa[t][j] = sb[t][j];
  }
  for (int j = 0; j < 64; ++j) hid[(long)g * 64 + j] = f2bf(sa[t][j]);
}

__global__ __launch_bounds__(256) void f32_to_bf16_k(const float* __restrict__ in,
                                                     __bf16* __restrict__ out, int nElem) {
  int i = blockIdx.x * 256 + threadIdx.x;
  if (i < nElem) out[i] = f2bf(in[i]);
}

// ---------------------------------------------------------------------------
// WMMA fragment helpers (wave32, 16x16x32 bf16):
//  A (16x32): lane L: m=L&15, hi=L>>4; vgpr p holds K=2p+hi*16, +1
//  B (32x16): lane L: n=L&15, K=(L>>4)*16 .. +15 (contiguous)  -> staged
//             transposed in LDS so this is one aligned 32B read
//  C/D: element r -> row = r + (L>>4)*8, col = L&15
// ---------------------------------------------------------------------------

enum { EPI_S1 = 0, EPI_S2 = 1, EPI_A2 = 2, EPI_S4 = 3, EPI_S5 = 4 };

template<int EPI, bool AF32>
__global__ __launch_bounds__(256) void gemm_k(
    const void* __restrict__ Av, int lda, long aStride /*elements per batch*/,
    const __bf16* __restrict__ B, int ldb,
    int N, int K,
    void* __restrict__ Outv, const float* __restrict__ bias) {
  __shared__ alignas(16) __bf16 lsA[16 * 32];
  __shared__ alignas(32) __bf16 lsBt[128 * 32];   // transposed [col][k]
  const int tid = threadIdx.x, lane = tid & 31, wid = tid >> 5;
  const int z = blockIdx.z;
  const long m0 = (long)blockIdx.y * 16;
  const int n0 = blockIdx.x * 128;
  const int NT = min(N - n0, 128);
  const int nt = wid * 16;
  const bool active = nt < NT;                     // wave-uniform

  const float*  Af = (const float*)Av  + (size_t)z * (size_t)aStride;
  const __bf16* Ab = (const __bf16*)Av + (size_t)z * (size_t)aStride;

  v8f acc = {};
  const int m = lane & 15, hi = lane >> 4;

  for (int k0 = 0; k0 < K; k0 += 32) {
    // ---- stage A tile (16x32) ----
#ifdef HAVE_ASYNC_COPY
    if (!AF32) {
      if (tid < 64) {                              // 64 lanes x 16B = 1KB tile
        int r = tid >> 2, seg = (tid & 3) * 8;     // seg in bf16 elements
        async_copy16(Ab + (m0 + r) * (long)lda + k0 + seg, &lsA[r * 32 + seg]);
      }
    } else
#endif
    {
      for (int idx = tid; idx < 512; idx += 256) {
        int r = idx >> 5, kk = idx & 31;
        long ai = (m0 + r) * (long)lda + (k0 + kk);
        lsA[idx] = AF32 ? f2bf(Af[ai]) : Ab[ai];
      }
    }
    // ---- stage B tile (32x128), transposed into [col][k] ----
    for (int idx = tid; idx < 4096; idx += 256) {
      int r = idx >> 7, c = idx & 127;
      __bf16 v = (n0 + c < N) ? B[(long)(k0 + r) * ldb + (n0 + c)] : f2bf(0.f);
      lsBt[c * 32 + r] = v;
    }
    // prefetch next K-step's B rows into cache while we compute
    if (k0 + 32 < K && tid < 32 && n0 + (lane & 7) * 16 < N)
      __builtin_prefetch(&B[(long)(k0 + 32 + (lane >> 3) * 8) * ldb + n0 + (lane & 7) * 16], 0, 1);
    async_wait0();
    __syncthreads();

    if (active) {
      v16bf af;
#pragma unroll
      for (int p = 0; p < 8; ++p) {
        int kk = 2 * p + hi * 16;
        af[2 * p]     = lsA[m * 32 + kk];
        af[2 * p + 1] = lsA[m * 32 + kk + 1];
      }
      v16bf bfr = *(const v16bf*)&lsBt[(nt + m) * 32 + hi * 16];
      acc = __builtin_amdgcn_wmma_f32_16x16x32_bf16(false, af, false, bfr,
                                                    (short)0, acc, false, false);
    }
    __syncthreads();
  }
  if (!active) return;

  const int colT = nt + m;
#pragma unroll
  for (int r = 0; r < 8; ++r) {
    long row = m0 + hi * 8 + r;
    int  col = n0 + colT;
    float v = acc[r];
    if (EPI == EPI_S1) {
      // rows = (img,h) of x; cols = 2n+c  ->  ws1[img][n][2h+c]
      long img = row >> 7; int hh = (int)(row & 127);
      int n = col >> 1, c = col & 1;
      ((__bf16*)Outv)[img * 20480 + (long)n * 256 + 2 * hh + c] = f2bf(v);
    } else if (EPI == EPI_S2) {
      // rows = n; cols = 2h'+c; batch z = b*64+i -> ws2[f=h'*80+n][b][2i+c]
      int n = (int)row; int hp = col >> 1, c = col & 1;
      long f = (long)hp * 80 + n;
      int b = z >> 6, ii = z & 63;
      ((__bf16*)Outv)[f * 2048 + b * 128 + 2 * ii + c] = f2bf(v);
    } else if (EPI == EPI_A2) {
      // rows = g; cols = channel c = o*64+i  ->  Wbuf[g][i][o]
      v += bias[col];
      int o = col >> 6, ii = col & 63;
      ((__bf16*)Outv)[row * 4096 + (long)ii * 64 + o] = f2bf(v);
    } else if (EPI == EPI_S4) {
      // rows = n; cols = 2h+c; batch z=(b,o) -> ws4[z][h][2n+c]
      int n = (int)row; int hh = col >> 1, c = col & 1;
      ((__bf16*)Outv)[(long)z * 20480 + (long)hh * 160 + 2 * n + c] = f2bf(v);
    } else {
      // S5: rows = h, cols = w, f32 out[z][h][w]
      ((float*)Outv)[(long)z * 16384 + row * 128 + col] = v;
    }
  }
}

// ---------------------------------------------------------------------------
// Per-frequency channel contraction: out[b,o] = sum_i x_ft[b,i] * W[i,o]
// (complex). One block per frequency f=(h',n). Expanded complex weight matrix
// BxT[2o+c][2i+a] is built in LDS from bf16 Wr/Wi (pure bit ops: sign flip is
// XOR 0x8000), then a single real [16x128]@[128x128] WMMA GEMM produces the
// interleaved complex result. x_ft tile is staged with async global->LDS.
// ---------------------------------------------------------------------------
__global__ __launch_bounds__(256) void einsum_k(const __bf16* __restrict__ ws2,
                                                const __bf16* __restrict__ Wr,
                                                const __bf16* __restrict__ Wi,
                                                __bf16* __restrict__ ws3) {
  const int f = blockIdx.x;                 // 0..10239
  const int hp = f / 80, n = f % 80;
  const int tid = threadIdx.x, lane = tid & 31, wid = tid >> 5;
  if (n >= 65) {                            // padded freqs: zero the slice
    for (int idx = tid; idx < 2048; idx += 256) {
      int b = idx >> 7, rem = idx & 127, o = rem >> 1, c = rem & 1;
      ws3[(long)(b * 64 + o) * 20480 + (long)n * 256 + 2 * hp + c] = f2bf(0.f);
    }
    return;
  }
  const long g = (long)hp * 65 + n;
  __shared__ alignas(16) __bf16 lsA[16 * 128];
  __shared__ alignas(32) __bf16 lsBt[128 * 128];  // [col=2o+c][K=2i+a]
  // stage x_ft tile [16][128] (4KB): one async b128 per thread
#ifdef HAVE_ASYNC_COPY
  async_copy16(ws2 + (long)f * 2048 + tid * 8, &lsA[tid * 8]);
#else
  for (int idx = tid; idx < 2048; idx += 256) lsA[idx] = ws2[(long)f * 2048 + idx];
#endif
  // build expanded complex weights with bit ops only
  const unsigned short* wrh = (const unsigned short*)(Wr + g * 4096);
  const unsigned short* wih = (const unsigned short*)(Wi + g * 4096);
  unsigned short* bt = (unsigned short*)lsBt;
  for (int idx = tid; idx < 16384; idx += 256) {
    int j = idx >> 7, K = idx & 127;
    int o = j >> 1, c = j & 1, k = K >> 1, a = K & 1;
    unsigned short hv;
    if (a == c) hv = wrh[k * 64 + o];
    else        hv = (unsigned short)(wih[k * 64 + o] ^ (a ? 0x8000u : 0x0u));
    bt[idx] = hv;
  }
  async_wait0();
  __syncthreads();

  v8f acc = {};
  const int m = lane & 15, hi = lane >> 4;
#pragma unroll
  for (int ks = 0; ks < 4; ++ks) {
    v16bf af;
#pragma unroll
    for (int p = 0; p < 8; ++p) {
      int kk = ks * 32 + 2 * p + hi * 16;
      af[2 * p]     = lsA[m * 128 + kk];
      af[2 * p + 1] = lsA[m * 128 + kk + 1];
    }
    v16bf bfr = *(const v16bf*)&lsBt[(wid * 16 + m) * 128 + ks * 32 + hi * 16];
    acc = __builtin_amdgcn_wmma_f32_16x16x32_bf16(false, af, false, bfr,
                                                  (short)0, acc, false, false);
  }
  const int j = wid * 16 + m;               // output col = 2o+c
  const int o = j >> 1, c = j & 1;
#pragma unroll
  for (int r = 0; r < 8; ++r) {
    int b = hi * 8 + r;
    ws3[(long)(b * 64 + o) * 20480 + (long)n * 256 + 2 * hp + c] = f2bf(acc[r]);
  }
}

// ---------------------------------------------------------------------------
// Host side
// ---------------------------------------------------------------------------
extern "C" void kernel_launch(void* const* d_in, const int* in_sizes, int n_in,
                              void* d_out, int out_size, void* d_ws, size_t ws_size,
                              hipStream_t stream) {
  if (n_in < 17) return;

  // Input flattening: expected dict-insertion order [x, params_real(8), params_imag(8)],
  // but tolerate jax-tree sorted-key order as well.
  bool x_first = (in_sizes[0] == 16777216);
  const float* x = (const float*)d_in[x_first ? 0 : 16];
  int baseR = x_first ? 1 : 8;
  int baseI = x_first ? 9 : 0;
  // semantic order: 0 w_in, 1 b_in, 2 wh0, 3 wh1, 4 bh0, 5 bh1, 6 w_out, 7 b_out
  int map_ins[8] = { 0, 1, 2, 3, 4, 5, 6, 7 };
  int map_srt[8] = { 6, 2, 4, 5, 0, 1, 7, 3 }; // sorted leaves: bh0,bh1,b_in,b_out,wh0,wh1,w_in,w_out
  const int* mp = (in_sizes[baseR] == 128) ? map_ins : map_srt;
  const float* PR[8]; const float* PI[8];
  for (int k = 0; k < 8; ++k) {
    PR[k] = (const float*)d_in[baseR + mp[k]];
    PI[k] = (const float*)d_in[baseI + mp[k]];
  }

  // Workspace carve (bytes)
  const size_t OFF_CWF   = 0;           // 128*160 bf16
  const size_t OFF_CHF   = 40960;       // 256*256
  const size_t OFF_CHI   = 172032;      // 256*256
  const size_t OFF_B5    = 303104;      // 160*128
  const size_t OFF_HIDR  = 344064;      // 8320*64
  const size_t OFF_HIDI  = 1409024;
  const size_t OFF_WOR   = 2473984;     // 64*4096 bf16 (w_out real, bf16)
  const size_t OFF_WOI   = 2998272;
  const size_t OFF_WBR   = 3522560;     // 8320*4096 bf16
  const size_t OFF_WBI   = 71680000;
  const size_t OFF_WSA   = 139837440;   // 1024*80*256 bf16 (ws1, reused as ws3)
  const size_t OFF_WSB   = 181780480;   // 10240*16*128 bf16 (ws2, reused as ws4)
  const size_t NEED      = 223723520;
  if (ws_size < NEED) return;

  char* W = (char*)d_ws;
  __bf16* cwf  = (__bf16*)(W + OFF_CWF);
  __bf16* chf  = (__bf16*)(W + OFF_CHF);
  __bf16* chi  = (__bf16*)(W + OFF_CHI);
  __bf16* b5   = (__bf16*)(W + OFF_B5);
  __bf16* hidR = (__bf16*)(W + OFF_HIDR);
  __bf16* hidI = (__bf16*)(W + OFF_HIDI);
  __bf16* worB = (__bf16*)(W + OFF_WOR);
  __bf16* woiB = (__bf16*)(W + OFF_WOI);
  __bf16* WbR  = (__bf16*)(W + OFF_WBR);
  __bf16* WbI  = (__bf16*)(W + OFF_WBI);
  __bf16* ws1  = (__bf16*)(W + OFF_WSA);
  __bf16* ws2  = (__bf16*)(W + OFF_WSB);
  __bf16* ws3  = ws1;   // ws1 dead after S2
  __bf16* ws4  = ws2;   // ws2 dead after einsum

  // 1) constants
  gen_consts_k<<<672, 256, 0, stream>>>(cwf, chf, chi, b5);
  // 2) MLP hidden stacks (real / imag nets)
  mlp_hidden_k<<<130, 64, 0, stream>>>(PR[0], PR[1], PR[2], PR[3], PR[4], PR[5], hidR);
  mlp_hidden_k<<<130, 64, 0, stream>>>(PI[0], PI[1], PI[2], PI[3], PI[4], PI[5], hidI);
  // 3) w_out to bf16
  f32_to_bf16_k<<<1024, 256, 0, stream>>>(PR[6], worB, 262144);
  f32_to_bf16_k<<<1024, 256, 0, stream>>>(PI[6], woiB, 262144);
  // 4) final MLP layer (WMMA GEMM), epilogue transposes (o*64+i) -> [i][o]
  gemm_k<EPI_A2, false><<<dim3(32, 520, 1), 256, 0, stream>>>(hidR, 64, 0, worB, 4096, 4096, 64, WbR, PR[7]);
  gemm_k<EPI_A2, false><<<dim3(32, 520, 1), 256, 0, stream>>>(hidI, 64, 0, woiB, 4096, 4096, 64, WbI, PI[7]);
  // 5) rfft along W: [131072,128] @ [128,160]
  gemm_k<EPI_S1, true><<<dim3(2, 8192, 1), 256, 0, stream>>>(x, 128, 0, cwf, 160, 160, 128, ws1, nullptr);
  // 6) fft along H: batched [80,256] @ [256,256], 1024 images
  gemm_k<EPI_S2, false><<<dim3(2, 5, 1024), 256, 0, stream>>>(ws1, 256, 20480, chf, 256, 256, 256, ws2, nullptr);
  // 7) per-frequency channel contraction (complex, WMMA)
  einsum_k<<<10240, 256, 0, stream>>>(ws2, WbR, WbI, ws3);
  // 8) inverse fft along H
  gemm_k<EPI_S4, false><<<dim3(2, 5, 1024), 256, 0, stream>>>(ws3, 256, 20480, chi, 256, 256, 256, ws4, nullptr);
  // 9) inverse rfft along W, f32 output
  gemm_k<EPI_S5, false><<<dim3(1, 8, 1024), 256, 0, stream>>>(ws4, 160, 20480, b5, 128, 128, 160, d_out, nullptr);
}
